// ChildSumTreeLSTM_27754078667562
// MI455X (gfx1250) — matrix operand order repaired and checked by reading, not compiled, not run on hardware
//
#include <hip/hip_runtime.h>
#include <hip/hip_bf16.h>

// ---------------------------------------------------------------------------
// Child-Sum TreeLSTM, complete 8-ary tree, depth 6.  N = 299593 nodes.
// Level-synchronous bottom-up sweep; all GEMMs via v_wmma_f32_16x16x32_bf16.
// ---------------------------------------------------------------------------

typedef __bf16 bf16_t;
typedef __attribute__((ext_vector_type(16))) __bf16 v16bf;
typedef __attribute__((ext_vector_type(8)))  __bf16 v8bf;
typedef __attribute__((ext_vector_type(8)))  float  v8f;

#define TREE_N    299593
#define LEAF_BASE 37449     // off(6)
#define LEAF_CNT  262144    // 8^6

__device__ __forceinline__ int imin(int a, int b) { return a < b ? a : b; }

__device__ __forceinline__ float sigmoidf_(float x) {
  return 1.0f / (1.0f + __expf(-x));
}

__device__ __forceinline__ v8f splat8(float x) {
  v8f v;
#pragma unroll
  for (int i = 0; i < 8; ++i) v[i] = x;
  return v;
}

__device__ __forceinline__ v8f wmma_bf16(v16bf a, v16bf b, v8f c) {
  // D = A(16x32 bf16) * B(32x16 bf16) + C(16x16 f32)
  return __builtin_amdgcn_wmma_f32_16x16x32_bf16(
      /*neg_a=*/false, a, /*neg_b=*/false, b,
      /*c_mod=*/(short)0, c, /*reuse_a=*/false, /*reuse_b=*/false);
}

// A-fragment (16x32, MxK): this lane's row pointer, elems 0..7 = k0..k0+7,
// elems 8..15 = k0+16..k0+23 (k0 = kstep*32 + (lane>>4)*8).  f32 src -> bf16.
__device__ __forceinline__ v16bf a_frag_f32(const float* __restrict__ row, int k0) {
  v16bf a;
#pragma unroll
  for (int i = 0; i < 8; ++i) {
    a[i]     = (bf16_t)row[k0 + i];
    a[i + 8] = (bf16_t)row[k0 + 16 + i];
  }
  return a;
}

__device__ __forceinline__ v16bf a_frag_bf16(const bf16_t* row, int k0) {
  union { v16bf v; v8bf h[2]; } u;
  u.h[0] = *(const v8bf*)(row + k0);
  u.h[1] = *(const v8bf*)(row + k0 + 16);
  return u.v;
}

// B-fragment (32x16, KxN): lane holds column n = t*16 + (lane&15), elems 0..7 =
// k0..k0+7, elems 8..15 = k0+16..k0+23 (k0 = kstep*32 + (lane>>4)*8).
// W is row-major [out_dim][128] so B[k][n] = W[n][k] -> no transpose needed.
__device__ __forceinline__ v16bf b_frag(const bf16_t* W, int n, int k0) {
  union { v16bf v; v8bf h[2]; } u;
  u.h[0] = *(const v8bf*)(W + (size_t)n * 128 + k0);
  u.h[1] = *(const v8bf*)(W + (size_t)n * 128 + k0 + 16);
  return u.v;
}

// ---------------------------------------------------------------------------
__global__ void f32_to_bf16_kernel(const float* __restrict__ src,
                                   bf16_t* __restrict__ dst, int n) {
  int i = blockIdx.x * blockDim.x + threadIdx.x;
  if (i < n) dst[i] = (bf16_t)src[i];
}

// ---------------------------------------------------------------------------
// Leaf level: h_sum = 0, fc_sum = 0.
// iou = x @ W_ioux^T + (b_ioux + b_iouh);  c = sig(i)*tanh(u);  h = sig(o)*tanh(c)
// One wave per 16-node tile, 8 tiles per 256-thread block.  16384 tiles total.
// ---------------------------------------------------------------------------
__global__ __launch_bounds__(256) void leaf_kernel(
    const float* __restrict__ x,
    const float* __restrict__ b_ioux, const float* __restrict__ b_iouh,
    const bf16_t* __restrict__ Wioux,
    float* __restrict__ c_all, bf16_t* __restrict__ h_bf) {
  const int lane = threadIdx.x & 31;
  const int wave = threadIdx.x >> 5;
  const int tile = blockIdx.x * 8 + wave;   // 0 .. 16383
  const int n15  = lane & 15;
  const int kh   = lane >> 4;

  const float* xrow = x + (size_t)(LEAF_BASE + tile * 16 + n15) * 128;
  v16bf xa[4];
#pragma unroll
  for (int ks = 0; ks < 4; ++ks) xa[ks] = a_frag_f32(xrow, ks * 32 + kh * 8);

#pragma unroll
  for (int t = 0; t < 8; ++t) {
    const int col = t * 16 + n15;
    v8f ai = splat8(b_ioux[col]       + b_iouh[col]);
    v8f ao = splat8(b_ioux[col + 128] + b_iouh[col + 128]);
    v8f au = splat8(b_ioux[col + 256] + b_iouh[col + 256]);
#pragma unroll
    for (int ks = 0; ks < 4; ++ks) {
      const int k0 = ks * 32 + kh * 8;
      ai = wmma_bf16(xa[ks], b_frag(Wioux, col, k0), ai);
      ao = wmma_bf16(xa[ks], b_frag(Wioux, col + 128, k0), ao);
      au = wmma_bf16(xa[ks], b_frag(Wioux, col + 256, k0), au);
    }
#pragma unroll
    for (int r = 0; r < 8; ++r) {
      const float cv = sigmoidf_(ai[r]) * tanhf(au[r]);
      const float hv = sigmoidf_(ao[r]) * tanhf(cv);
      const size_t node = (size_t)LEAF_BASE + tile * 16 + kh * 8 + r;
      c_all[node * 128 + col] = cv;
      h_bf [node * 128 + col] = (bf16_t)hv;
    }
  }
}

// ---------------------------------------------------------------------------
// Internal levels.  Per 16-node tile (one wave):
//  1) h_sum = sum_b h_ch[b]   (f32 accum, bf16 into per-wave LDS tile)
//  2) fx    = x @ W_fx^T + b_fx + b_fh                        (8 N-tiles)
//  3) per child b: f = sig(fx + h_b @ W_fh^T);  fc += f * c_b (W_fh in LDS)
//  4) iou = x@W_ioux^T + h_sum@W_iouh^T + biases;  combine with fc
// ---------------------------------------------------------------------------
__global__ __launch_bounds__(256) void inner_kernel(
    const float* __restrict__ x,
    const float* __restrict__ b_ioux, const float* __restrict__ b_iouh,
    const float* __restrict__ b_fx,   const float* __restrict__ b_fh,
    const bf16_t* __restrict__ Wioux, const bf16_t* __restrict__ Wiouh,
    const bf16_t* __restrict__ Wfx,   const bf16_t* __restrict__ Wfh,
    float* __restrict__ c_all, bf16_t* __restrict__ h_bf,
    int lvl_base, int child_base, int n_l, int ntiles,
    int is_root, float* __restrict__ out) {
  __shared__ __align__(32) bf16_t lds_wfh[128 * 128];     // 32 KB
  __shared__ __align__(32) bf16_t lds_hsum[8][16 * 128];  // 32 KB (4 KB / wave)

  // Stage W_fh (bf16, 32768 B) into LDS cooperatively.
  {
    const uint4* src = (const uint4*)Wfx;  (void)src;
    const uint4* s = (const uint4*)Wfh;
    uint4* d = (uint4*)lds_wfh;
#pragma unroll
    for (int i = 0; i < 8; ++i) d[threadIdx.x + i * 256] = s[threadIdx.x + i * 256];
  }
  __syncthreads();

  const int lane = threadIdx.x & 31;
  const int wave = threadIdx.x >> 5;
  const int tile = blockIdx.x * 8 + wave;
  if (tile >= ntiles) return;

  const int n15 = lane & 15;
  const int kh  = lane >> 4;
  const int ncl = n_l - 1;

  // --- 1) h_sum tile into LDS (2 lanes per row, 64 cols each) ---------------
  {
    const int row = lane >> 1;
    const int cb  = (lane & 1) * 64;
    const int nloc = imin(tile * 16 + row, ncl);
#pragma unroll
    for (int cc = 0; cc < 64; cc += 8) {
      float acc[8];
#pragma unroll
      for (int j = 0; j < 8; ++j) acc[j] = 0.0f;
      for (int b = 0; b < 8; ++b) {
        const size_t ch = (size_t)child_base + (size_t)nloc * 8 + b;
        const v8bf hv = *(const v8bf*)(h_bf + ch * 128 + cb + cc);
#pragma unroll
        for (int j = 0; j < 8; ++j) acc[j] += (float)hv[j];
      }
      v8bf o;
#pragma unroll
      for (int j = 0; j < 8; ++j) o[j] = (bf16_t)acc[j];
      *(v8bf*)&lds_hsum[wave][row * 128 + cb + cc] = o;
    }
  }
  __builtin_amdgcn_wave_barrier();  // in-wave DS ordering fence for cross-lane reuse

  // --- 2) x fragments + fx pre-activation -----------------------------------
  const float* xrow = x + (size_t)(lvl_base + imin(tile * 16 + n15, ncl)) * 128;
  v16bf xa[4];
#pragma unroll
  for (int ks = 0; ks < 4; ++ks) xa[ks] = a_frag_f32(xrow, ks * 32 + kh * 8);

  v8f accfx[8];
#pragma unroll
  for (int t = 0; t < 8; ++t) {
    const int col = t * 16 + n15;
    accfx[t] = splat8(b_fx[col] + b_fh[col]);
#pragma unroll
    for (int ks = 0; ks < 4; ++ks)
      accfx[t] = wmma_bf16(xa[ks], b_frag(Wfx, col, ks * 32 + kh * 8), accfx[t]);
  }

  // --- 3) children: forget gates + fc_sum -----------------------------------
  v8f fc[8];
#pragma unroll
  for (int t = 0; t < 8; ++t) fc[t] = splat8(0.0f);

  for (int b = 0; b < 8; ++b) {
    const size_t chA = (size_t)child_base + (size_t)imin(tile * 16 + n15, ncl) * 8 + b;
    const bf16_t* hrow = h_bf + chA * 128;
    v16bf ha[4];
#pragma unroll
    for (int ks = 0; ks < 4; ++ks) ha[ks] = a_frag_bf16(hrow, ks * 32 + kh * 8);

#pragma unroll
    for (int t = 0; t < 8; ++t) {
      const int col = t * 16 + n15;
      v8f acc = accfx[t];
#pragma unroll
      for (int ks = 0; ks < 4; ++ks)
        acc = wmma_bf16(ha[ks], b_frag(lds_wfh, col, ks * 32 + kh * 8), acc);
#pragma unroll
      for (int r = 0; r < 8; ++r) {
        const int mr = kh * 8 + r;
        const size_t chc =
            (size_t)child_base + (size_t)imin(tile * 16 + mr, ncl) * 8 + b;
        const float cv = c_all[chc * 128 + col];
        fc[t][r] += sigmoidf_(acc[r]) * cv;
      }
    }
  }

  // --- 4) iou GEMMs + combine ------------------------------------------------
  v16bf hsa[4];
#pragma unroll
  for (int ks = 0; ks < 4; ++ks)
    hsa[ks] = a_frag_bf16(&lds_hsum[wave][n15 * 128], ks * 32 + kh * 8);

#pragma unroll
  for (int t = 0; t < 8; ++t) {
    const int col = t * 16 + n15;
    v8f ai = splat8(b_ioux[col]       + b_iouh[col]);
    v8f ao = splat8(b_ioux[col + 128] + b_iouh[col + 128]);
    v8f au = splat8(b_ioux[col + 256] + b_iouh[col + 256]);
#pragma unroll
    for (int ks = 0; ks < 4; ++ks) {
      const int k0 = ks * 32 + kh * 8;
      ai = wmma_bf16(xa[ks],  b_frag(Wioux, col, k0), ai);
      ai = wmma_bf16(hsa[ks], b_frag(Wiouh, col, k0), ai);
      ao = wmma_bf16(xa[ks],  b_frag(Wioux, col + 128, k0), ao);
      ao = wmma_bf16(hsa[ks], b_frag(Wiouh, col + 128, k0), ao);
      au = wmma_bf16(xa[ks],  b_frag(Wioux, col + 256, k0), au);
      au = wmma_bf16(hsa[ks], b_frag(Wiouh, col + 256, k0), au);
    }
#pragma unroll
    for (int r = 0; r < 8; ++r) {
      const int mr   = kh * 8 + r;
      const int node = tile * 16 + mr;
      const float cv = sigmoidf_(ai[r]) * tanhf(au[r]) + fc[t][r];
      const float hv = sigmoidf_(ao[r]) * tanhf(cv);
      if (node < n_l) {
        const size_t g = (size_t)(lvl_base + node) * 128 + col;
        c_all[g] = cv;
        h_bf[g]  = (bf16_t)hv;
        if (is_root && node == 0) { out[col] = cv; out[128 + col] = hv; }
      }
    }
  }
}

// ---------------------------------------------------------------------------
extern "C" void kernel_launch(void* const* d_in, const int* in_sizes, int n_in,
                              void* d_out, int out_size, void* d_ws, size_t ws_size,
                              hipStream_t stream) {
  const float* x       = (const float*)d_in[0];
  const float* W_ioux  = (const float*)d_in[1];
  const float* b_ioux  = (const float*)d_in[2];
  const float* W_iouh  = (const float*)d_in[3];
  const float* b_iouh  = (const float*)d_in[4];
  const float* W_fx    = (const float*)d_in[5];
  const float* b_fx    = (const float*)d_in[6];
  const float* W_fh    = (const float*)d_in[7];
  const float* b_fh    = (const float*)d_in[8];
  float* out = (float*)d_out;

  // Workspace layout (all regions 256B-aligned by construction).
  char* ws = (char*)d_ws;
  size_t off = 0;
  float*  c_all = (float*)(ws + off);  off += (size_t)TREE_N * 128 * 4;  // 153.4 MB
  bf16_t* h_bf  = (bf16_t*)(ws + off); off += (size_t)TREE_N * 128 * 2;  //  76.7 MB
  bf16_t* Wioux_bf = (bf16_t*)(ws + off); off += 384 * 128 * 2;
  bf16_t* Wiouh_bf = (bf16_t*)(ws + off); off += 384 * 128 * 2;
  bf16_t* Wfx_bf   = (bf16_t*)(ws + off); off += 128 * 128 * 2;
  bf16_t* Wfh_bf   = (bf16_t*)(ws + off); off += 128 * 128 * 2;

  // Weight conversion to bf16 (tiny; stays L2-resident afterwards).
  f32_to_bf16_kernel<<<(384 * 128 + 255) / 256, 256, 0, stream>>>(W_ioux, Wioux_bf, 384 * 128);
  f32_to_bf16_kernel<<<(384 * 128 + 255) / 256, 256, 0, stream>>>(W_iouh, Wiouh_bf, 384 * 128);
  f32_to_bf16_kernel<<<(128 * 128 + 255) / 256, 256, 0, stream>>>(W_fx, Wfx_bf, 128 * 128);
  f32_to_bf16_kernel<<<(128 * 128 + 255) / 256, 256, 0, stream>>>(W_fh, Wfh_bf, 128 * 128);

  // Leaf level: 262144 nodes -> 16384 tiles -> 2048 blocks of 8 waves.
  leaf_kernel<<<LEAF_CNT / 16 / 8, 256, 0, stream>>>(x, b_ioux, b_iouh, Wioux_bf,
                                                     c_all, h_bf);

  // Internal levels 5..0 (BFS offsets off(l) = (8^l - 1)/7).
  static const int lvl_off[8] = {0, 1, 9, 73, 585, 4681, 37449, 299593};
  for (int l = 5; l >= 0; --l) {
    int n_l = 1;
    for (int i = 0; i < l; ++i) n_l *= 8;
    const int ntiles = (n_l + 15) / 16;
    const int blocks = (ntiles + 7) / 8;
    inner_kernel<<<blocks, 256, 0, stream>>>(
        x, b_ioux, b_iouh, b_fx, b_fh,
        Wioux_bf, Wiouh_bf, Wfx_bf, Wfh_bf,
        c_all, h_bf,
        lvl_off[l], lvl_off[l + 1], n_l, ntiles,
        (l == 0) ? 1 : 0, out);
  }
}
